// VectorQuantizerEMA_73126113181953
// MI455X (gfx1250) — compile-verified, matrix operand order checked by reading, not run on hardware
//
#include <hip/hip_runtime.h>
#include <hip/hip_bf16.h>

// ---------------------------------------------------------------------------
// VQ-VAE EMA forward for gfx1250 (MI455X), wave32 + WMMA bf16 16x16x32.
// Round 3: fix async-to-LDS builtin arg types (vV4i*1, vV4i*3 per diagnostic).
// LDS-resident codebook (async-to-LDS), 2 row-tiles per wave (2 WMMAs per B
// fragment), XOR-swizzled LDS chunks.
// ---------------------------------------------------------------------------

typedef __attribute__((ext_vector_type(16))) __bf16 v16bf;
typedef __attribute__((ext_vector_type(8)))  float  v8f;

union Frag { uint4 u[2]; v16bf v; };

#if __has_builtin(__builtin_amdgcn_global_load_async_to_lds_b128) && \
    __has_builtin(__builtin_amdgcn_s_wait_asynccnt)
#define VQ_HAS_ASYNC 1
typedef int vi4 __attribute__((vector_size(16)));          // matches builtin's V4i
typedef __attribute__((address_space(1))) vi4 as1_vi4;     // global
typedef __attribute__((address_space(3))) vi4 as3_vi4;     // LDS
#else
#define VQ_HAS_ASYNC 0
#endif

__device__ __forceinline__ unsigned short f2bf(float x) {
    union { float f; unsigned u; } v; v.f = x;
    unsigned r = v.u + 0x7FFFu + ((v.u >> 16) & 1u);   // round-to-nearest-even
    return (unsigned short)(r >> 16);
}

// Read one 16 B chunk from a 128 B-row LDS tile with XOR bank swizzle.
__device__ __forceinline__ uint4 lds_chunk(const char* base, int row, int j) {
    int jj = j ^ (row & 7);
    return *(const uint4*)(base + row * 128 + jj * 16);
}

// ---- kernel 1: codebook fp32 -> bf16 + 0.5*||e_k||^2 ----------------------
__global__ __launch_bounds__(256) void vq_prep_kernel(
    const float* __restrict__ ew, unsigned short* __restrict__ cb,
    float* __restrict__ half_norm)
{
    int k = blockIdx.x * 256 + threadIdx.x;
    if (k >= 512) return;
    float s = 0.0f;
    for (int d = 0; d < 64; ++d) {
        float v = ew[k * 64 + d];
        s += v * v;
        cb[k * 64 + d] = f2bf(v);
    }
    half_norm[k] = 0.5f * s;
}

// ---- kernel 2: zero accumulators (every launch; harness doesn't re-init) --
__global__ void vq_zero_kernel(float* __restrict__ acc, float* __restrict__ sse)
{
    int i = blockIdx.x * 256 + threadIdx.x;
    if (i < 33280) acc[i] = 0.0f;
    if (i == 0)    sse[0] = 0.0f;
}

// ---- kernel 3: WMMA distance argmin ---------------------------------------
// Block: 128 threads = 4 wave32. Each wave owns 32 rows (two 16-row tiles),
// block covers 128 rows (never crosses a batch: 128 | 1024).
// Score S[n,k] = x_n . e_k - 0.5||e_k||^2 ; argmax_k == argmin_k dist.
// LDS layout (dynamic, 84 KB):
//   [0,      65536)  bf16 codebook, 512 rows x 128 B, chunks XOR-swizzled
//   [65536,  67584)  half_norm f32[512]
//   [67584,  83968)  bf16 A tile, 128 rows x 128 B, chunks XOR-swizzled
__global__ __launch_bounds__(128) void vq_argmin_kernel(
    const float* __restrict__ x,              // BCHW [64,64,32,32]
    const unsigned short* __restrict__ cb,    // bf16 codebook [512][64] (global)
    const float* __restrict__ half_norm,      // [512]
    int*   __restrict__ idx_out,              // [65536]
    float* __restrict__ idx_f_out)            // [65536] (float copy for d_out)
{
    extern __shared__ char smem[];
    char*  scb = smem;                         // codebook tile
    float* shn = (float*)(smem + 65536);       // half norms
    char*  sA  = smem + 67584;                 // A tile

    const int tid  = threadIdx.x;
    const int lane = tid & 31;
    const int wave = tid >> 5;
    const int hi   = lane >> 4;                // K-half per ISA 7.12.2
    const int col  = lane & 15;

    const int nbase = blockIdx.x * 128;
    const int b = nbase >> 10;                 // batch
    const int p = nbase & 1023;                // h*W+w base within batch

    // ---- stage codebook (64 KB) + norms (2 KB) into LDS -------------------
#if VQ_HAS_ASYNC
    {   // one b128 per lane per call; ASYNCcnt tracks completion
        const uint4* g = (const uint4*)cb;
        for (int i = 0; i < 32; ++i) {
            int c = i * 128 + tid;             // 4096 chunks total
            int k = c >> 3, j = c & 7;
            const uint4* src = g + c;
            char* dst = scb + (size_t)k * 128 + (size_t)(j ^ (k & 7)) * 16;
            __builtin_amdgcn_global_load_async_to_lds_b128(
                (as1_vi4*)(uintptr_t)src,
                (as3_vi4*)(unsigned)(uintptr_t)dst, 0, 0);
        }
        const uint4* hsrc = (const uint4*)half_norm + tid;   // 128 chunks
        __builtin_amdgcn_global_load_async_to_lds_b128(
            (as1_vi4*)(uintptr_t)hsrc,
            (as3_vi4*)(unsigned)(uintptr_t)((uint4*)shn + tid), 0, 0);
    }
#else
    {
        const uint4* g = (const uint4*)cb;
        for (int i = 0; i < 32; ++i) {
            int c = i * 128 + tid;
            int k = c >> 3, j = c & 7;
            *(uint4*)(scb + (size_t)k * 128 + (size_t)(j ^ (k & 7)) * 16) = g[c];
        }
        ((uint4*)shn)[tid] = ((const uint4*)half_norm)[tid];
    }
#endif

    // ---- stage A tile: 128 rows x 64 dims, fp32 (d-stride 1024) -> bf16 ---
    #pragma unroll 4
    for (int i = 0; i < 64; ++i) {
        int e = i * 128 + tid;                 // 8192 elements
        int d = e >> 7, r = e & 127;           // consecutive tid -> coalesced r
        float xv = x[((size_t)(b * 64 + d) << 10) + (p + r)];
        int off = r * 128 + (((d >> 3) ^ (r & 7)) << 4) + ((d & 7) << 1);
        *(unsigned short*)(sA + off) = f2bf(xv);
    }

#if VQ_HAS_ASYNC
    __builtin_amdgcn_s_wait_asynccnt(0);       // async copies landed in LDS
#endif
    __syncthreads();

    // ---- A fragments: 2 tiles x 2 d-blocks (ISA 7.12.2 16-bit A 16x32) ----
    Frag a00, a01, a10, a11;
    {
        int r0 = wave * 32 + col;              // tile 0 row
        int r1 = r0 + 16;                      // tile 1 row
        a00.u[0] = lds_chunk(sA, r0, 0 + hi);  a00.u[1] = lds_chunk(sA, r0, 2 + hi);
        a01.u[0] = lds_chunk(sA, r0, 4 + hi);  a01.u[1] = lds_chunk(sA, r0, 6 + hi);
        a10.u[0] = lds_chunk(sA, r1, 0 + hi);  a10.u[1] = lds_chunk(sA, r1, 2 + hi);
        a11.u[0] = lds_chunk(sA, r1, 4 + hi);  a11.u[1] = lds_chunk(sA, r1, 6 + hi);
    }

    float best0[8], best1[8]; int bidx0[8], bidx1[8];
    #pragma unroll
    for (int j = 0; j < 8; ++j) {
        best0[j] = -3.4e38f; bidx0[j] = 0;
        best1[j] = -3.4e38f; bidx1[j] = 0;
    }

    // ---- hot loop: pure LDS loads + 4 WMMAs per k-tile --------------------
    #pragma unroll 2
    for (int kt = 0; kt < 32; ++kt) {
        int k = kt * 16 + col;                 // this lane's C column (codebook row)
        Frag b0, b1;
        b0.u[0] = lds_chunk(scb, k, 0 + hi);   b0.u[1] = lds_chunk(scb, k, 2 + hi);
        b1.u[0] = lds_chunk(scb, k, 4 + hi);   b1.u[1] = lds_chunk(scb, k, 6 + hi);
        float hn = shn[k];

        v8f c0 = {}, c1 = {};
        // interleave the two accumulator chains to hide WMMA->WMMA RAW hazard
        c0 = __builtin_amdgcn_wmma_f32_16x16x32_bf16(false, a00.v, false, b0.v,
                                                     (short)0, c0, false, false);
        c1 = __builtin_amdgcn_wmma_f32_16x16x32_bf16(false, a10.v, false, b0.v,
                                                     (short)0, c1, false, false);
        c0 = __builtin_amdgcn_wmma_f32_16x16x32_bf16(false, a01.v, false, b1.v,
                                                     (short)0, c0, false, false);
        c1 = __builtin_amdgcn_wmma_f32_16x16x32_bf16(false, a11.v, false, b1.v,
                                                     (short)0, c1, false, false);
        #pragma unroll
        for (int j = 0; j < 8; ++j) {
            float v0 = c0[j] - hn;
            if (v0 > best0[j]) { best0[j] = v0; bidx0[j] = k; }  // lowest k on ties
            float v1 = c1[j] - hn;
            if (v1 > best1[j]) { best1[j] = v1; bidx1[j] = k; }
        }
    }

    // ---- reduce across the 16 lanes holding the same rows -----------------
    #pragma unroll
    for (int m = 1; m < 16; m <<= 1) {
        #pragma unroll
        for (int j = 0; j < 8; ++j) {
            float ov = __shfl_xor(best0[j], m, 32);
            int   oi = __shfl_xor(bidx0[j], m, 32);
            if (ov > best0[j] || (ov == best0[j] && oi < bidx0[j])) {
                best0[j] = ov; bidx0[j] = oi;
            }
            float ov1 = __shfl_xor(best1[j], m, 32);
            int   oi1 = __shfl_xor(bidx1[j], m, 32);
            if (ov1 > best1[j] || (ov1 == best1[j] && oi1 < bidx1[j])) {
                best1[j] = ov1; bidx1[j] = oi1;
            }
        }
    }
    if (col == 0) {                            // lanes 0 (rows 0-7) / 16 (rows 8-15)
        #pragma unroll
        for (int j = 0; j < 8; ++j) {
            int n0 = nbase + wave * 32 + hi * 8 + j;
            idx_out[n0]   = bidx0[j];
            idx_f_out[n0] = (float)bidx0[j];
            int n1 = n0 + 16;
            idx_out[n1]   = bidx1[j];
            idx_f_out[n1] = (float)bidx1[j];
        }
    }
}

// ---- kernel 4: counts + dw segment sums (LDS f32 atomics, sparse flush) ---
__global__ __launch_bounds__(256) void vq_stats_kernel(
    const float* __restrict__ x, const int* __restrict__ idx,
    float* __restrict__ acc /* [512 counts | 512*64 dw] */)
{
    extern __shared__ float lacc[];            // 33280 floats = 130 KB (< 320 KB WGP LDS)
    const int tid = threadIdx.x;
    for (int i = tid; i < 33280; i += 256) lacc[i] = 0.0f;
    __syncthreads();

    const int bb = blockIdx.x;                 // one batch image per block
    const int nb = bb << 10;
    for (int i = tid; i < 1024; i += 256)
        atomicAdd(&lacc[idx[nb + i]], 1.0f);
    for (int i = tid; i < 65536; i += 256) {   // (d, r): consecutive tid -> coalesced r
        int d = i >> 10, r = i & 1023;
        int k = idx[nb + r];
        float xv = x[((size_t)(bb * 64 + d) << 10) + r];
        atomicAdd(&lacc[512 + k * 64 + d], xv);
    }
    __syncthreads();
    for (int i = tid; i < 33280; i += 256) {
        float v = lacc[i];
        if (v != 0.0f) atomicAdd(&acc[i], v);
    }
}

// ---- kernel 5: EMA finalize -> updated codebook ---------------------------
__global__ __launch_bounds__(512) void vq_finalize_kernel(
    const float* __restrict__ ema_cs, const float* __restrict__ ema_w,
    const float* __restrict__ acc, float* __restrict__ eu)
{
    __shared__ float red[512];
    int k = threadIdx.x;                       // 0..511
    float ncs = 0.99f * ema_cs[k] + 0.01f * acc[k];
    red[k] = ncs;
    __syncthreads();
    for (int s = 256; s > 0; s >>= 1) {
        if (k < s) red[k] += red[k + s];
        __syncthreads();
    }
    float n  = red[0];
    float cs = (ncs + 1e-5f) / (n + 512.0f * 1e-5f) * n;
    float inv = 1.0f / cs;
    for (int d = 0; d < 64; ++d) {
        float w = 0.99f * ema_w[k * 64 + d] + 0.01f * acc[512 + k * 64 + d];
        eu[k * 64 + d] = w * inv;
    }
}

// ---- kernel 6: gather quantized, write BCHW output, accumulate SSE --------
__global__ __launch_bounds__(256) void vq_output_kernel(
    const float* __restrict__ x, const int* __restrict__ idx,
    const float* __restrict__ eu, float* __restrict__ out,
    float* __restrict__ sse)
{
    __shared__ float red[256];
    const int tid = threadIdx.x;
    const int nbase = blockIdx.x * 64;
    const int b = nbase >> 10;
    const int p = nbase & 1023;
    float a = 0.0f;
    #pragma unroll
    for (int i = 0; i < 16; ++i) {
        int e = i * 256 + tid;
        int d = e >> 6, r = e & 63;
        int n = nbase + r;
        size_t off = ((size_t)(b * 64 + d) << 10) + (p + r);
        float q  = eu[idx[n] * 64 + d];        // updated codebook (matches reference)
        float xv = x[off];
        out[off] = q;                          // straight-through fwd == quantized
        float dd = q - xv;
        a += dd * dd;
    }
    red[tid] = a;
    __syncthreads();
    for (int s = 128; s > 0; s >>= 1) {
        if (tid < s) red[tid] += red[tid + s];
        __syncthreads();
    }
    if (tid == 0) atomicAdd(sse, red[0]);
}

__global__ void vq_loss_kernel(const float* __restrict__ sse,
                               float* __restrict__ loss_out)
{
    if (threadIdx.x == 0 && blockIdx.x == 0)
        loss_out[0] = 0.25f * sse[0] / 4194304.0f;  // COMMITMENT_COST * mean
}

// ---------------------------------------------------------------------------
extern "C" void kernel_launch(void* const* d_in, const int* in_sizes, int n_in,
                              void* d_out, int out_size, void* d_ws, size_t ws_size,
                              hipStream_t stream) {
    (void)in_sizes; (void)n_in; (void)out_size; (void)ws_size;
    const float* x       = (const float*)d_in[0];   // [64,64,32,32]
    const float* embed_w = (const float*)d_in[1];   // [512,64]
    const float* ema_cs  = (const float*)d_in[2];   // [512]
    const float* ema_w   = (const float*)d_in[3];   // [512,64]

    float* out = (float*)d_out;                     // out | loss | indices
    float* loss_out = out + 4194304;
    float* idx_f    = out + 4194305;

    char* ws = (char*)d_ws;                         // ~594 KB total
    unsigned short* cb        = (unsigned short*)(ws);            //  64 KB bf16 codebook
    float*          half_norm = (float*)(ws + 65536);             //   2 KB
    int*            idxb      = (int*)  (ws + 67584);             // 256 KB
    float*          acc       = (float*)(ws + 329728);            // 130 KB counts+dw
    float*          eu        = (float*)(ws + 462848);            // 128 KB updated codebook
    float*          sse       = (float*)(ws + 593920);            //   4 B

    vq_prep_kernel    <<<2,    256, 0, stream>>>(embed_w, cb, half_norm);
    vq_zero_kernel    <<<131,  256, 0, stream>>>(acc, sse);
    vq_argmin_kernel  <<<512,  128, 83968, stream>>>(x, cb, half_norm, idxb, idx_f);
    vq_stats_kernel   <<<64,   256, 33280 * sizeof(float), stream>>>(x, idxb, acc);
    vq_finalize_kernel<<<1,    512, 0, stream>>>(ema_cs, ema_w, acc, eu);
    vq_output_kernel  <<<1024, 256, 0, stream>>>(x, idxb, eu, out, sse);
    vq_loss_kernel    <<<1,    64,  0, stream>>>(sse, loss_out);
}
